// GCNModel_61512521614082
// MI455X (gfx1250) — matrix operand order repaired and checked by reading, not compiled, not run on hardware
//
#include <hip/hip_runtime.h>
#include <hip/hip_bf16.h>

// ---------------------------------------------------------------------------
// Problem dims (fixed by the reference)
// ---------------------------------------------------------------------------
#define BB    128      // graphs / batch
#define LL    128      // seq len
#define NNODE (BB*LL)  // 16384 nodes
#define EE    262144   // edges
#define EMBD  300
#define EMBP  320      // EMB padded to multiple of 32 (zero pad)
#define HH    256      // LSTM hidden per dir
#define G4    1024     // 4*H gate width
#define FF    512      // feature width (2H and GCN)
#define KCHUNK 256     // K-panel staged in LDS per buffer (64 x 256 f16 = 32KB)

typedef _Float16 v16h __attribute__((ext_vector_type(16)));
typedef _Float16 v8h  __attribute__((ext_vector_type(8)));
typedef float    v8f  __attribute__((ext_vector_type(8)));
typedef unsigned int u32x4 __attribute__((ext_vector_type(4)));
typedef int          i32x4 __attribute__((ext_vector_type(4)));
typedef int          i32x8 __attribute__((ext_vector_type(8)));

#if defined(__has_builtin)
#  if __has_builtin(__builtin_amdgcn_tensor_load_to_lds)
#    define HAVE_TDM 1
#  endif
#endif
#ifndef HAVE_TDM
#  define HAVE_TDM 0
#endif

__device__ __forceinline__ float sig_(float x) { return 1.f / (1.f + __expf(-x)); }

// ---------------------------------------------------------------------------
// Issue (async) stage of a 64-row x `chunk`-col f16 panel of Bt into LDS.
// TDM path: wave 0 issues one tensor_load_to_lds (D# per ISA cdna5 §8:
// group0 = {count=1, lds_addr, global_addr, type=2}; group1 = {data_size=2B,
// huge tensor dims (no OOB taken), tile = chunk x 64, dim0 stride = ldB}).
// Completion is waited separately (stage_wait) so DMA overlaps compute.
// Fallback: cooperative 16B copies (ds_store_b128), synchronous.
// ---------------------------------------------------------------------------
__device__ __forceinline__ void stage_issue(_Float16* sB, const _Float16* Bp,
                                            int ldB, int n0, int kc, int chunk,
                                            int wv, int tid)
{
#if HAVE_TDM
    if (wv == 0) {
        unsigned long long ga = (unsigned long long)(uintptr_t)(Bp + (size_t)n0 * ldB + kc);
        unsigned int lds = (unsigned int)(uintptr_t)sB;     // flat LDS addr, low 32 bits
        u32x4 g0;
        g0[0] = 1u;                                         // count=1, user mode, no gather
        g0[1] = lds;                                        // lds_addr (bytes)
        g0[2] = (unsigned int)ga;                           // global_addr[31:0]
        g0[3] = (unsigned int)((ga >> 32) & 0x01FFFFFFu)    // global_addr[56:32]
              | (2u << 30);                                 // type = 2 ("image")
        unsigned long long td0 = 1ull << 30;                // tensor_dim0 (huge, no OOB)
        unsigned long long td1 = 1ull << 20;                // tensor_dim1 (huge)
        unsigned long long st0 = (unsigned long long)(unsigned int)ldB; // dim0 stride
        unsigned long long q0 = (1ull << 16) | (td0 << 48); // data_size=1 (2B) | td0[15:0]
        unsigned long long q1 = (td0 >> 16) | (td1 << 16)
                              | ((unsigned long long)(unsigned int)chunk << 48); // tile_dim0
        unsigned long long q2 = 64ull                       // tile_dim1 = 64 rows
                              | ((st0 & 0xFFFFFFFFull) << 32); // stride[31:0]
        unsigned long long q3 = (st0 >> 32) & 0xFFFFull;    // stride[47:32]; dim1_stride=0
        i32x8 g1;
        g1[0] = (int)(unsigned int)q0;  g1[1] = (int)(unsigned int)(q0 >> 32);
        g1[2] = (int)(unsigned int)q1;  g1[3] = (int)(unsigned int)(q1 >> 32);
        g1[4] = (int)(unsigned int)q2;  g1[5] = (int)(unsigned int)(q2 >> 32);
        g1[6] = (int)(unsigned int)q3;  g1[7] = (int)(unsigned int)(q3 >> 32);
        i32x4 z4 = {0, 0, 0, 0};
#if __has_include(<hip/amd_detail/amd_gfx1250_TDM.h>)
        i32x8 z8 = {0, 0, 0, 0, 0, 0, 0, 0};
        __builtin_amdgcn_tensor_load_to_lds(g0, g1, z4, z4, z8, 0);
#else
        __builtin_amdgcn_tensor_load_to_lds(g0, g1, z4, z4, 0);
#endif
    }
#else
    const int vpr = chunk >> 3;            // 16B vectors per row
    const int total = 64 * vpr;
    for (int idx = tid; idx < total; idx += 256) {
        int r = idx / vpr, v = idx - r * vpr;
        *(v8h*)(sB + (size_t)r * chunk + v * 8) =
            *(const v8h*)(Bp + (size_t)(n0 + r) * ldB + kc + v * 8);
    }
#endif
    (void)tid;
}

__device__ __forceinline__ void stage_wait(int wv)
{
#if HAVE_TDM
    if (wv == 0) {
#  if __has_builtin(__builtin_amdgcn_s_wait_tensorcnt)
        __builtin_amdgcn_s_wait_tensorcnt(0);
#  else
        asm volatile("s_wait_tensorcnt 0x0" ::: "memory");
#  endif
    }
#else
    (void)wv;
#endif
}

// ---------------------------------------------------------------------------
// WMMA GEMM:  C[M][ldc] = A[M][lda](f16) * Bt[N][ldb]^T(f16) (+ A2*B2t^T) + init
//   A row-major [M][K] (K-contiguous). Bt is B stored transposed [N][K]
//   (K-contiguous). The 64-col B panel is staged in LDS via the Tensor Data
//   Mover with DOUBLE BUFFERING: chunk i+1's DMA is issued before computing
//   chunk i, and s_wait_tensorcnt happens only after the compute -> DMA
//   overlaps WMMA. All 8 waves share each panel (8x reuse via ds_load_b128).
//   A fragments are two global_load_b128 per lane, each lane half holding a
//   contiguous 16-k run (k = 16*hi + s), same order as B. init: Cinit matrix OR bias[N] OR 0.
//   Each wave computes a 16(M) x 64(N) strip; block = 8 waves = 128 rows.
//   All launch shapes are exact multiples (no partial blocks -> barriers safe).
// ---------------------------------------------------------------------------
__global__ void __launch_bounds__(256)
wmma_gemm_kernel(const _Float16* __restrict__ A,  int lda,
                 const _Float16* __restrict__ Bt, int ldb, int K,
                 const _Float16* __restrict__ A2, int lda2,
                 const _Float16* __restrict__ B2t,int ldb2, int K2,
                 const float* __restrict__ bias,
                 const float* __restrict__ Cinit, long long ldci,
                 float* __restrict__ C, int ldc, int M)
{
    __shared__ _Float16 sBuf[2][64 * KCHUNK];   // 2 x 32 KB double buffer
    const int tid  = threadIdx.x;
    const int lane = tid & 31;
    const int wv   = tid >> 5;
    const int m0   = blockIdx.y * 128 + wv * 16;
    const int n0   = blockIdx.x * 64;
    const int lm   = lane & 15;            // M index within tile (A), N index (B/C)
    const int hi   = lane >> 4;            // half-select per ISA layout

    v8f acc[4];
    if (Cinit) {
        #pragma unroll
        for (int j = 0; j < 4; ++j) {
            #pragma unroll
            for (int r = 0; r < 8; ++r)
                acc[j][r] = Cinit[(long long)(m0 + r + hi * 8) * ldci
                                  + (n0 + j * 16 + lm)];
        }
    } else {
        #pragma unroll
        for (int j = 0; j < 4; ++j) {
            float bv = bias ? bias[n0 + j * 16 + lm] : 0.f;
            #pragma unroll
            for (int r = 0; r < 8; ++r) acc[j][r] = bv;
        }
    }

    const int npass = A2 ? 2 : 1;
    for (int pass = 0; pass < npass; ++pass) {
        const _Float16* Ap = pass ? A2  : A;
        const _Float16* Bp = pass ? B2t : Bt;
        const int ldA = pass ? lda2 : lda;
        const int ldB = pass ? ldb2 : ldb;
        const int Kp  = pass ? K2   : K;
        const _Float16* arow = Ap + (long long)(m0 + lm) * ldA + hi * 16;
        const int nchunk = (Kp + KCHUNK - 1) / KCHUNK;

        // prologue: stage chunk 0 (synchronous)
        __syncthreads();                               // prior consumers done with LDS
        {
            const int c0 = (Kp < KCHUNK) ? Kp : KCHUNK;
            stage_issue(sBuf[0], Bp, ldB, n0, 0, c0, wv, tid);
            stage_wait(wv);
        }
        __syncthreads();

        for (int ci = 0; ci < nchunk; ++ci) {
            const int kc    = ci * KCHUNK;
            const int chunk = (Kp - kc < KCHUNK) ? (Kp - kc) : KCHUNK;
            const _Float16* cur = sBuf[ci & 1];
            const bool more = (ci + 1 < nchunk);
            if (more) {                                // async DMA of next chunk
                const int kn = kc + KCHUNK;
                const int cn = (Kp - kn < KCHUNK) ? (Kp - kn) : KCHUNK;
                stage_issue(sBuf[(ci & 1) ^ 1], Bp, ldB, n0, kn, cn, wv, tid);
            }

            auto kbody = [&](int kst, int stride) {
                __builtin_prefetch(arow + kc + kst + 64, 0, 0);  // global_prefetch_b8
                v8h alo = *(const v8h*)(arow + kc + kst);        // K = kb .. kb+7
                v8h ahi = *(const v8h*)(arow + kc + kst + 8);    // K = kb+8 .. kb+15
                v16h a = __builtin_shufflevector(alo, ahi,
                            0,1,2,3,4,5,6,7,8,9,10,11,12,13,14,15);
                const int kb = kst + hi * 16;
                // load all four B fragments first, then issue the four WMMAs
                v16h b0 = *(const v16h*)(cur + (size_t)(     lm) * stride + kb);
                v16h b1 = *(const v16h*)(cur + (size_t)(16 + lm) * stride + kb);
                v16h b2 = *(const v16h*)(cur + (size_t)(32 + lm) * stride + kb);
                v16h b3 = *(const v16h*)(cur + (size_t)(48 + lm) * stride + kb);
                acc[0] = __builtin_amdgcn_wmma_f32_16x16x32_f16(
                            false, a, false, b0, (short)0, acc[0], false, false);
                acc[1] = __builtin_amdgcn_wmma_f32_16x16x32_f16(
                            false, a, false, b1, (short)0, acc[1], false, false);
                acc[2] = __builtin_amdgcn_wmma_f32_16x16x32_f16(
                            false, a, false, b2, (short)0, acc[2], false, false);
                acc[3] = __builtin_amdgcn_wmma_f32_16x16x32_f16(
                            false, a, false, b3, (short)0, acc[3], false, false);
            };
            if (chunk == KCHUNK) {                     // constant trip count: unroll+pipeline
                #pragma unroll 4
                for (int kst = 0; kst < KCHUNK; kst += 32) kbody(kst, KCHUNK);
            } else {
                for (int kst = 0; kst < chunk; kst += 32) kbody(kst, chunk);
            }

            if (more) stage_wait(wv);                  // DMA done before publishing
            __syncthreads();
        }
    }

    #pragma unroll
    for (int j = 0; j < 4; ++j) {
        #pragma unroll
        for (int r = 0; r < 8; ++r)
            C[(long long)(m0 + r + hi * 8) * ldc + (n0 + j * 16 + lm)] = acc[j][r];
    }
}

// ---------------------------------------------------------------------------
// Elementwise / data-movement kernels
// ---------------------------------------------------------------------------
__global__ void zero_kernel(float* __restrict__ p, long long n) {
    long long i = (long long)blockIdx.x * blockDim.x + threadIdx.x;
    if (i < n) p[i] = 0.f;
}

// xe16[m][k] = (f16) embed[word[m]][k]  (k < 300, zero-padded to 320)
__global__ void embed_gather_kernel(const int* __restrict__ word,
                                    const float* __restrict__ table,
                                    _Float16* __restrict__ xe16)
{
    long long i = (long long)blockIdx.x * blockDim.x + threadIdx.x;
    if (i >= (long long)NNODE * EMBP) return;
    int m = (int)(i / EMBP), k = (int)(i % EMBP);
    float v = (k < EMBD) ? table[(long long)word[m] * EMBD + k] : 0.f;
    xe16[i] = (_Float16)v;
}

// out[r][k<Kpad] = (f16) in[r][k<Kin], zero padded
__global__ void pad_cvt_kernel(const float* __restrict__ in,
                               _Float16* __restrict__ out,
                               int Kin, int Kpad, long long total)
{
    long long i = (long long)blockIdx.x * blockDim.x + threadIdx.x;
    if (i >= total) return;
    int r = (int)(i / Kpad), k = (int)(i % Kpad);
    out[i] = (_Float16)((k < Kin) ? in[(long long)r * Kin + k] : 0.f);
}

// 512x512 transpose + convert: out[n][k] = (f16) in[k][n]
__global__ void tr_cvt_kernel(const float* __restrict__ in,
                              _Float16* __restrict__ out)
{
    long long i = (long long)blockIdx.x * blockDim.x + threadIdx.x;
    if (i >= (long long)FF * FF) return;
    int n = (int)(i / FF), k = (int)(i % FF);
    out[i] = (_Float16)in[(long long)k * FF + n];
}

__global__ void cvt16_kernel(const float* __restrict__ x,
                             _Float16* __restrict__ y, long long n)
{
    long long i = (long long)blockIdx.x * blockDim.x + threadIdx.x;
    if (i < n) y[i] = (_Float16)x[i];
}

// LSTM pointwise update; grid.z = dir (0 fwd, 1 bwd); 32768 threads/dir
__global__ void lstm_update_kernel(const float* __restrict__ gates,
                                   float* __restrict__ cst,
                                   _Float16* __restrict__ h16,
                                   float* __restrict__ x0, int t)
{
    const int dir = blockIdx.z;
    const int i = blockIdx.x * blockDim.x + threadIdx.x;   // 0..BB*HH-1
    const int b = i >> 8, j = i & (HH - 1);
    const float* g = gates + (long long)dir * (BB * G4) + (long long)b * G4;
    float iv = sig_(g[j]);
    float fv = sig_(g[HH + j]);
    float gv = tanhf(g[2 * HH + j]);
    float ov = sig_(g[3 * HH + j]);
    float* cp = cst + (long long)dir * (BB * HH) + i;
    float c = fv * (*cp) + iv * gv;
    *cp = c;
    float h = ov * tanhf(c);
    h16[(long long)dir * (BB * HH) + i] = (_Float16)h;
    int tt = dir ? (LL - 1 - t) : t;
    x0[(long long)(b * LL + tt) * FF + dir * HH + j] = h;
}

__global__ void scatter_cnt_kernel(const int* __restrict__ dst,
                                   float* __restrict__ cnt, int E)
{
    int e = blockIdx.x * blockDim.x + threadIdx.x;
    if (e < E) atomicAdd(&cnt[dst[e]], 1.0f);
}

// one block per edge; agg[dst] += x[src]  (L2-resident f32 atomics)
__global__ void scatter_add_kernel(const float* __restrict__ x,
                                   const int* __restrict__ src,
                                   const int* __restrict__ dst,
                                   float* __restrict__ agg)
{
    const int e = blockIdx.x;
    const int s = src[e], d = dst[e];
    const float* xr = x + (long long)s * FF;
    float* ar = agg + (long long)d * FF;
    for (int c = threadIdx.x; c < FF; c += blockDim.x)
        atomicAdd(&ar[c], xr[c]);
}

// agg16 = (f16)(agg / max(cnt,1))
__global__ void div_cvt_kernel(const float* __restrict__ agg,
                               const float* __restrict__ cnt,
                               _Float16* __restrict__ agg16)
{
    long long i = (long long)blockIdx.x * blockDim.x + threadIdx.x;
    if (i >= (long long)NNODE * FF) return;
    float c = cnt[i / FF];
    c = c < 1.f ? 1.f : c;
    agg16[i] = (_Float16)(agg[i] / c);
}

// per-row L2 normalize + ReLU (in place) + write f16 copy for next GEMM
__global__ void __launch_bounds__(256)
norm_relu_kernel(float* __restrict__ x, _Float16* __restrict__ x16)
{
    const int n = blockIdx.x;
    float* row = x + (long long)n * FF;
    float ss = 0.f;
    for (int c = threadIdx.x; c < FF; c += 256) { float v = row[c]; ss += v * v; }
    for (int o = 16; o > 0; o >>= 1) ss += __shfl_down(ss, o);
    __shared__ float red[8];
    __shared__ float invs;
    if ((threadIdx.x & 31) == 0) red[threadIdx.x >> 5] = ss;
    __syncthreads();
    if (threadIdx.x == 0) {
        float t = 0.f;
        for (int i = 0; i < 8; ++i) t += red[i];
        invs = 1.f / fmaxf(sqrtf(t), 1e-12f);
    }
    __syncthreads();
    const float inv = invs;
    for (int c = threadIdx.x; c < FF; c += 256) {
        float v = row[c] * inv;
        v = v > 0.f ? v : 0.f;
        row[c] = v;
        x16[(long long)n * FF + c] = (_Float16)v;
    }
}

// mean-pool 128 nodes per graph, then 512x2 classifier
__global__ void __launch_bounds__(256)
pool_final_kernel(const float* __restrict__ x, const float* __restrict__ Wlin,
                  const float* __restrict__ blin, float* __restrict__ out)
{
    __shared__ float pooled[FF];
    const int b = blockIdx.x;
    for (int k = threadIdx.x; k < FF; k += 256) {
        float s = 0.f;
        const float* base = x + (long long)(b * LL) * FF + k;
        for (int nloc = 0; nloc < LL; ++nloc) s += base[(long long)nloc * FF];
        pooled[k] = s * (1.f / (float)LL);
    }
    __syncthreads();
    if (threadIdx.x < 2) {
        float s = blin[threadIdx.x];
        for (int k = 0; k < FF; ++k) s += pooled[k] * Wlin[k * 2 + threadIdx.x];
        out[b * 2 + threadIdx.x] = s;
    }
}

// ---------------------------------------------------------------------------
// Host orchestration
// ---------------------------------------------------------------------------
extern "C" void kernel_launch(void* const* d_in, const int* in_sizes, int n_in,
                              void* d_out, int out_size, void* d_ws, size_t ws_size,
                              hipStream_t stream)
{
    const int*   word  = (const int*)  d_in[0];
    const int*   eidx  = (const int*)  d_in[1];
    const float* table = (const float*)d_in[3];
    const float* wih_f = (const float*)d_in[4];
    const float* whh_f = (const float*)d_in[5];
    const float* b_f   = (const float*)d_in[6];
    const float* wih_b = (const float*)d_in[7];
    const float* whh_b = (const float*)d_in[8];
    const float* b_b   = (const float*)d_in[9];
    const float* Wl[3] = {(const float*)d_in[10], (const float*)d_in[13], (const float*)d_in[16]};
    const float* Wr[3] = {(const float*)d_in[11], (const float*)d_in[14], (const float*)d_in[17]};
    const float* bl[3] = {(const float*)d_in[12], (const float*)d_in[15], (const float*)d_in[18]};
    const float* Wlin  = (const float*)d_in[19];
    const float* blin  = (const float*)d_in[20];
    const int* src = eidx;
    const int* dst = eidx + EE;

    // workspace carve (256B aligned)
    size_t off = 0;
    auto carve = [&](size_t bytes) -> void* {
        off = (off + 255) & ~(size_t)255;
        void* p = (char*)d_ws + off;
        off += bytes;
        return p;
    };
    _Float16* xe16  = (_Float16*)carve((size_t)NNODE * EMBP * 2);
    _Float16* wf16  = (_Float16*)carve((size_t)G4 * EMBP * 2);   // wih_f as Bt [1024][320]
    _Float16* wb16  = (_Float16*)carve((size_t)G4 * EMBP * 2);
    _Float16* uf16  = (_Float16*)carve((size_t)G4 * HH * 2);     // whh_f as Bt [1024][256]
    _Float16* ub16  = (_Float16*)carve((size_t)G4 * HH * 2);
    float*    xs_f  = (float*)   carve((size_t)NNODE * G4 * 4);
    float*    xs_b  = (float*)   carve((size_t)NNODE * G4 * 4);
    float*    gates = (float*)   carve((size_t)2 * BB * G4 * 4);
    float*    cst   = (float*)   carve((size_t)2 * BB * HH * 4);
    _Float16* h16   = (_Float16*)carve((size_t)2 * BB * HH * 2);
    float*    x0    = (float*)   carve((size_t)NNODE * FF * 4);
    float*    x1    = (float*)   carve((size_t)NNODE * FF * 4);
    _Float16* x16   = (_Float16*)carve((size_t)NNODE * FF * 2);
    float*    agg   = (float*)   carve((size_t)NNODE * FF * 4);
    _Float16* agg16 = (_Float16*)carve((size_t)NNODE * FF * 2);
    float*    cnt   = (float*)   carve((size_t)NNODE * 4);
    _Float16* Wt[6];
    for (int i = 0; i < 6; ++i) Wt[i] = (_Float16*)carve((size_t)FF * FF * 2);

    const dim3 blk(256);
    auto g1 = [](long long n) { return dim3((unsigned)((n + 255) / 256)); };

    // ---- weight / input conversion -------------------------------------
    embed_gather_kernel<<<g1((long long)NNODE * EMBP), blk, 0, stream>>>(word, table, xe16);
    pad_cvt_kernel<<<g1((long long)G4 * EMBP), blk, 0, stream>>>(wih_f, wf16, EMBD, EMBP, (long long)G4 * EMBP);
    pad_cvt_kernel<<<g1((long long)G4 * EMBP), blk, 0, stream>>>(wih_b, wb16, EMBD, EMBP, (long long)G4 * EMBP);
    pad_cvt_kernel<<<g1((long long)G4 * HH),   blk, 0, stream>>>(whh_f, uf16, HH, HH, (long long)G4 * HH);
    pad_cvt_kernel<<<g1((long long)G4 * HH),   blk, 0, stream>>>(whh_b, ub16, HH, HH, (long long)G4 * HH);
    for (int l = 0; l < 3; ++l) {
        tr_cvt_kernel<<<g1((long long)FF * FF), blk, 0, stream>>>(Wl[l], Wt[2 * l]);
        tr_cvt_kernel<<<g1((long long)FF * FF), blk, 0, stream>>>(Wr[l], Wt[2 * l + 1]);
    }
    zero_kernel<<<g1(2LL * BB * HH), blk, 0, stream>>>(cst, 2LL * BB * HH);
    zero_kernel<<<g1((2LL * BB * HH) / 2), blk, 0, stream>>>((float*)h16, (2LL * BB * HH) / 2);

    // ---- BiLSTM input projection (big WMMA GEMMs) ----------------------
    wmma_gemm_kernel<<<dim3(G4 / 64, NNODE / 128), blk, 0, stream>>>(
        xe16, EMBP, wf16, EMBP, EMBP,
        nullptr, 0, nullptr, 0, 0,
        b_f, nullptr, 0, xs_f, G4, NNODE);
    wmma_gemm_kernel<<<dim3(G4 / 64, NNODE / 128), blk, 0, stream>>>(
        xe16, EMBP, wb16, EMBP, EMBP,
        nullptr, 0, nullptr, 0, 0,
        b_b, nullptr, 0, xs_b, G4, NNODE);

    // ---- BiLSTM recurrence: 128 steps, WMMA per step -------------------
    for (int s = 0; s < LL; ++s) {
        wmma_gemm_kernel<<<dim3(G4 / 64, 1), blk, 0, stream>>>(
            h16, HH, uf16, HH, HH,
            nullptr, 0, nullptr, 0, 0,
            nullptr, xs_f + (size_t)s * G4, (long long)LL * G4,
            gates, G4, BB);
        wmma_gemm_kernel<<<dim3(G4 / 64, 1), blk, 0, stream>>>(
            h16 + BB * HH, HH, ub16, HH, HH,
            nullptr, 0, nullptr, 0, 0,
            nullptr, xs_b + (size_t)(LL - 1 - s) * G4, (long long)LL * G4,
            gates + BB * G4, G4, BB);
        lstm_update_kernel<<<dim3(BB, 1, 2), blk, 0, stream>>>(gates, cst, h16, x0, s);
    }

    // ---- GraphSAGE x3 ---------------------------------------------------
    cvt16_kernel<<<g1((long long)NNODE * FF), blk, 0, stream>>>(x0, x16, (long long)NNODE * FF);
    float* bufs[2] = { x0, x1 };
    for (int l = 0; l < 3; ++l) {
        float* xin  = bufs[l & 1];
        float* xout = bufs[(l & 1) ^ 1];
        zero_kernel<<<g1((long long)NNODE * FF), blk, 0, stream>>>(agg, (long long)NNODE * FF);
        zero_kernel<<<g1(NNODE), blk, 0, stream>>>(cnt, NNODE);
        scatter_cnt_kernel<<<g1(EE), blk, 0, stream>>>(dst, cnt, EE);
        scatter_add_kernel<<<dim3(EE), blk, 0, stream>>>(xin, src, dst, agg);
        div_cvt_kernel<<<g1((long long)NNODE * FF), blk, 0, stream>>>(agg, cnt, agg16);
        wmma_gemm_kernel<<<dim3(FF / 64, NNODE / 128), blk, 0, stream>>>(
            agg16, FF, Wt[2 * l], FF, FF,
            x16,   FF, Wt[2 * l + 1], FF, FF,
            bl[l], nullptr, 0, xout, FF, NNODE);
        norm_relu_kernel<<<dim3(NNODE), blk, 0, stream>>>(xout, x16);
    }

    // ---- pool + classifier ---------------------------------------------
    pool_final_kernel<<<dim3(BB), blk, 0, stream>>>(x1, Wlin, blin, (float*)d_out);

    (void)in_sizes; (void)n_in; (void)out_size; (void)ws_size;
}
